// vPFAE_GEN_68539088110348
// MI455X (gfx1250) — matrix operand order repaired and checked by reading, not compiled, not run on hardware
//
#include <hip/hip_runtime.h>
#include <hip/hip_bf16.h>

typedef __attribute__((ext_vector_type(16))) _Float16 v16h;
typedef __attribute__((ext_vector_type(8)))  _Float16 v8h;
typedef __attribute__((ext_vector_type(8)))  float    v8f;

#define K32MAX 96   // max padded K (96 for MLP1 second GEMM)
#define NOUTMAX 96  // max Nout (96 for MLP1 first GEMM)

// Issue one async cache->LDS dword copy (gfx1250, tracked by ASYNCcnt).
// Low 32 bits of a generic pointer to __shared__ == LDS byte offset.
__device__ __forceinline__ void async_copy_b32(const float* gsrc, float* ldst)
{
    unsigned loff = (unsigned)(size_t)ldst;
    unsigned long long ga = (unsigned long long)(size_t)gsrc;
    asm volatile("global_load_async_to_lds_b32 %0, %1, off"
                 :: "v"(loff), "v"(ga) : "memory");
}
__device__ __forceinline__ void async_wait0()
{
    asm volatile("s_wait_asynccnt 0x0" ::: "memory");
}

// ---------------------------------------------------------------------------
// Generic WMMA GEMM: C[M,Nout] = act(A[M,K] @ W[K,Nout] + bias)
// Requirements: Nout % 16 == 0, K % 4 == 0 (padded to mult of 32 in LDS).
// Block = 128 threads = 4 waves; each block: 64-row tile; wave -> 16 rows.
// ---------------------------------------------------------------------------
__global__ __launch_bounds__(128)
void gemm_wmma_f16(const float* __restrict__ A, const float* __restrict__ W,
                   const float* __restrict__ bias, float* __restrict__ C,
                   int M, int K, int Nout, int act)
{
    __shared__ __align__(32) _Float16 As[64 * K32MAX];
    __shared__ __align__(32) _Float16 Ws[NOUTMAX * K32MAX];

    const int tid = threadIdx.x;
    const int K32 = (K + 31) & ~31;
    const int rowBase = blockIdx.x * 64;

    // Stage W^T (col-major for B fragments) as f16, zero-pad K -> K32.
    for (int i = tid; i < Nout * K32; i += 128) {
        int n = i / K32, k = i - n * K32;
        Ws[n * K32MAX + k] = (k < K) ? (_Float16)W[(size_t)k * Nout + n]
                                     : (_Float16)0.f;
    }
    // Stage 64-row A tile as f16, zero-pad rows/K.
    for (int i = tid; i < 64 * K32; i += 128) {
        int r = i / K32, k = i - r * K32;
        int row = rowBase + r;
        As[r * K32MAX + k] = (row < M && k < K)
                               ? (_Float16)A[(size_t)row * K + k]
                               : (_Float16)0.f;
    }
    __syncthreads();

    const int wave = tid >> 5;
    const int lane = tid & 31;
    const int hs   = (lane >> 4) & 1;   // lane half-select
    const int l15  = lane & 15;
    const int rloc = wave * 16;         // this wave's 16-row sub-tile

    for (int nt = 0; nt < Nout; nt += 16) {
        v8f acc = {};
        for (int k0 = 0; k0 < K32; k0 += 32) {
            // A fragment: lane row = rloc+l15; elems e -> K = k0 + hs*16 + e
            // (contiguous k order, matching B)
            const _Float16* ap = &As[(rloc + l15) * K32MAX + k0 + hs * 16];
            union { v16h v; v8h h[2]; } au;
            au.h[0] = *(const v8h*)(ap);
            au.h[1] = *(const v8h*)(ap + 8);
            // B fragment: lane col = nt+l15; elems e -> K = k0 + hs*16 + e
            const _Float16* bp = &Ws[(nt + l15) * K32MAX + k0 + hs * 16];
            v16h b = *(const v16h*)(bp);
            acc = __builtin_amdgcn_wmma_f32_16x16x32_f16(
                      false, au.v, false, b, (short)0, acc, false, false);
        }
        // C/D layout: VGPR r -> row = hs*8 + r, col = l15
        #pragma unroll
        for (int r = 0; r < 8; ++r) {
            int row = rowBase + rloc + hs * 8 + r;
            int col = nt + l15;
            if (row < M) {
                float v = acc[r];
                if (bias) v += bias[col];
                if (act == 1) v = fmaxf(v, 0.f);
                C[(size_t)row * Nout + col] = v;
            }
        }
    }
}

// ---------------------------------------------------------------------------
// Fill kernels
// ---------------------------------------------------------------------------
__global__ void fill_f32(float* p, float v, size_t n) {
    size_t i = (size_t)blockIdx.x * blockDim.x + threadIdx.x;
    if (i < n) p[i] = v;
}
__global__ void fill_u32(unsigned* p, unsigned v, size_t n) {
    size_t i = (size_t)blockIdx.x * blockDim.x + threadIdx.x;
    if (i < n) p[i] = v;
}

// ---------------------------------------------------------------------------
// GENConv edge pass 1: per-dst per-channel max of msg = relu(h[src]+e)+EPS.
// Messages are strictly positive -> float max == int max on raw bits.
// Edge-linear weights are DMA'd to LDS via async-to-LDS (ASYNCcnt path);
// inner dot reads We[k*d+c] at wave-uniform addresses (LDS broadcast).
// ---------------------------------------------------------------------------
__global__ __launch_bounds__(256)
void edge_msg_max(const float* __restrict__ ea, const int* __restrict__ src,
                  const int* __restrict__ dst, const float* __restrict__ h,
                  const float* __restrict__ wE, const float* __restrict__ bE,
                  int* __restrict__ mmax, int E, int d)
{
    __shared__ float We[16 * 48];   // [k][c] verbatim copy of wE
    __shared__ float Be[48];
    const int tid = threadIdx.x;
    for (int i = tid; i < 16 * d; i += 256) async_copy_b32(wE + i, &We[i]);
    for (int i = tid; i < d;      i += 256) async_copy_b32(bE + i, &Be[i]);
    async_wait0();
    __syncthreads();

    int e = blockIdx.x * 256 + tid;
    if (e >= E) return;
    int s = src[e], t = dst[e];
    float ev[16];
    #pragma unroll
    for (int k = 0; k < 16; ++k) ev[k] = ea[(size_t)e * 16 + k];
    const float* hrow = &h[(size_t)s * d];
    int* mrow = &mmax[(size_t)t * d];
    for (int c = 0; c < d; ++c) {
        float a = Be[c];
        #pragma unroll
        for (int k = 0; k < 16; ++k) a += ev[k] * We[k * d + c];
        float m = fmaxf(hrow[c] + a, 0.f) + 1e-7f;
        atomicMax(&mrow[c], __float_as_int(m));
    }
}

// ---------------------------------------------------------------------------
// GENConv edge pass 2: ex = exp(msg - max[dst]); accumulate denom and ex*msg.
// ---------------------------------------------------------------------------
__global__ __launch_bounds__(256)
void edge_msg_sum(const float* __restrict__ ea, const int* __restrict__ src,
                  const int* __restrict__ dst, const float* __restrict__ h,
                  const float* __restrict__ wE, const float* __restrict__ bE,
                  const int* __restrict__ mmax, float* __restrict__ den,
                  float* __restrict__ num, int E, int d)
{
    __shared__ float We[16 * 48];
    __shared__ float Be[48];
    const int tid = threadIdx.x;
    for (int i = tid; i < 16 * d; i += 256) async_copy_b32(wE + i, &We[i]);
    for (int i = tid; i < d;      i += 256) async_copy_b32(bE + i, &Be[i]);
    async_wait0();
    __syncthreads();

    int e = blockIdx.x * 256 + tid;
    if (e >= E) return;
    int s = src[e], t = dst[e];
    float ev[16];
    #pragma unroll
    for (int k = 0; k < 16; ++k) ev[k] = ea[(size_t)e * 16 + k];
    const float* hrow = &h[(size_t)s * d];
    const int* mrow = &mmax[(size_t)t * d];
    float* drow = &den[(size_t)t * d];
    float* nrow = &num[(size_t)t * d];
    for (int c = 0; c < d; ++c) {
        float a = Be[c];
        #pragma unroll
        for (int k = 0; k < 16; ++k) a += ev[k] * We[k * d + c];
        float m  = fmaxf(hrow[c] + a, 0.f) + 1e-7f;
        float ex = __expf(m - __int_as_float(mrow[c]));
        atomicAdd(&drow[c], ex);
        atomicAdd(&nrow[c], ex * m);
    }
}

// out (in den buffer) = num/(den+1e-16) + h   (softmax aggregate + residual)
__global__ void combine_agg(float* den, const float* __restrict__ num,
                            const float* __restrict__ h, size_t total)
{
    size_t i = (size_t)blockIdx.x * blockDim.x + threadIdx.x;
    if (i < total) den[i] = num[i] / (den[i] + 1e-16f) + h[i];
}

// ---------------------------------------------------------------------------
// BatchNorm (training stats): block-partial sums in LDS, then global atomics.
// ---------------------------------------------------------------------------
__global__ __launch_bounds__(256)
void bn_reduce(const float* __restrict__ zt, float* __restrict__ sums,
               float* __restrict__ sqs, int N, int d)
{
    __shared__ float ls[96];
    __shared__ float lq[96];
    int tid = threadIdx.x;
    for (int i = tid; i < d; i += 256) { ls[i] = 0.f; lq[i] = 0.f; }
    __syncthreads();
    size_t total = (size_t)N * d;
    for (size_t i = (size_t)blockIdx.x * 256 + tid; i < total;
         i += (size_t)gridDim.x * 256) {
        float v = zt[i];
        int c = (int)(i % (size_t)d);
        atomicAdd(&ls[c], v);
        atomicAdd(&lq[c], v * v);
    }
    __syncthreads();
    for (int i = tid; i < d; i += 256) {
        atomicAdd(&sums[i], ls[i]);
        atomicAdd(&sqs[i],  lq[i]);
    }
}

__global__ void bn_apply_relu(float* zt, const float* __restrict__ sums,
                              const float* __restrict__ sqs,
                              const float* __restrict__ g,
                              const float* __restrict__ be, int N, int d)
{
    size_t i = (size_t)blockIdx.x * blockDim.x + threadIdx.x;
    if (i >= (size_t)N * d) return;
    int c = (int)(i % (size_t)d);
    float invN = 1.f / (float)N;
    float mean = sums[c] * invN;
    float var  = sqs[c] * invN - mean * mean;
    float v = (zt[i] - mean) * rsqrtf(var + 1e-5f) * g[c] + be[c];
    zt[i] = fmaxf(v, 0.f);
}

// ---------------------------------------------------------------------------
// GCN: degree, symmetric norm, self-loop init, edge scatter (both heads)
// ---------------------------------------------------------------------------
__global__ void gcn_deg(const int* __restrict__ dst, int* __restrict__ deg, int E)
{
    int e = blockIdx.x * blockDim.x + threadIdx.x;
    if (e < E) atomicAdd(&deg[dst[e]], 1);
}
__global__ void gcn_dinv(const int* __restrict__ deg, float* __restrict__ dinv, int N)
{
    int n = blockIdx.x * blockDim.x + threadIdx.x;
    if (n < N) dinv[n] = rsqrtf((float)(deg[n] + 1));   // +1 self loop
}
__global__ void gcn_init(const float* __restrict__ xwmu,
                         const float* __restrict__ xwls,
                         const float* __restrict__ dinv,
                         const float* __restrict__ bmu,
                         const float* __restrict__ bls,
                         float* __restrict__ omu, float* __restrict__ ols, int N)
{
    size_t i = (size_t)blockIdx.x * blockDim.x + threadIdx.x;
    if (i >= (size_t)N * 16) return;
    int n = (int)(i >> 4), c = (int)(i & 15);
    float d2 = dinv[n] * dinv[n];
    omu[i] = xwmu[i] * d2 + bmu[c];
    ols[i] = xwls[i] * d2 + bls[c];
}
__global__ __launch_bounds__(256)
void gcn_edge(const int* __restrict__ src, const int* __restrict__ dst,
              const float* __restrict__ dinv, const float* __restrict__ xwmu,
              const float* __restrict__ xwls, float* __restrict__ omu,
              float* __restrict__ ols, int E)
{
    int e = blockIdx.x * 256 + threadIdx.x;
    if (e >= E) return;
    int s = src[e], t = dst[e];
    float nm = dinv[s] * dinv[t];
    const float* ms = &xwmu[(size_t)s * 16];
    const float* ls = &xwls[(size_t)s * 16];
    float* mo = &omu[(size_t)t * 16];
    float* lo = &ols[(size_t)t * 16];
    #pragma unroll
    for (int c = 0; c < 16; ++c) {
        atomicAdd(&mo[c], ms[c] * nm);
        atomicAdd(&lo[c], ls[c] * nm);
    }
}

// ---------------------------------------------------------------------------
static inline int cdiv(long long a, long long b) { return (int)((a + b - 1) / b); }

extern "C" void kernel_launch(void* const* d_in, const int* in_sizes, int n_in,
                              void* d_out, int out_size, void* d_ws, size_t ws_size,
                              hipStream_t stream)
{
    const float* x    = (const float*)d_in[0];
    const int*   ei   = (const int*)  d_in[1];
    const float* ea   = (const float*)d_in[2];
    const float* w_src1 = (const float*)d_in[3];  const float* b_src1 = (const float*)d_in[4];
    const float* w_edg1 = (const float*)d_in[5];  const float* b_edg1 = (const float*)d_in[6];
    const float* w_m1a  = (const float*)d_in[7];  const float* b_m1a  = (const float*)d_in[8];
    const float* g1     = (const float*)d_in[9];  const float* be1    = (const float*)d_in[10];
    const float* w_m1b  = (const float*)d_in[11]; const float* b_m1b  = (const float*)d_in[12];
    const float* w_src2 = (const float*)d_in[13]; const float* b_src2 = (const float*)d_in[14];
    const float* w_edg2 = (const float*)d_in[15]; const float* b_edg2 = (const float*)d_in[16];
    const float* w_m2a  = (const float*)d_in[17]; const float* b_m2a  = (const float*)d_in[18];
    const float* g2     = (const float*)d_in[19]; const float* be2    = (const float*)d_in[20];
    const float* w_m2b  = (const float*)d_in[21]; const float* b_m2b  = (const float*)d_in[22];
    const float* w_mu   = (const float*)d_in[23]; const float* b_mu   = (const float*)d_in[24];
    const float* w_ls   = (const float*)d_in[25]; const float* b_ls   = (const float*)d_in[26];

    const int N = in_sizes[0] / 64;
    const int E = in_sizes[2] / 16;
    const int* srcI = ei;
    const int* dstI = ei + E;

    // Workspace carve-up (floats)
    float* ws = (float*)d_ws;
    float* hsrc = ws;            ws += (size_t)N * 48;  // lin_src output
    int*   mmax = (int*)ws;      ws += (size_t)N * 48;  // per-dst channel max (bits)
    float* den  = ws;            ws += (size_t)N * 48;  // softmax denom -> combined out
    float* num  = ws;            ws += (size_t)N * 48;  // softmax numerator
    float* zt   = ws;            ws += (size_t)N * 96;  // MLP hidden
    float* hmid = ws;            ws += (size_t)N * 48;  // layer1 output
    float* hfin = ws;            ws += (size_t)N * 32;  // layer2 output
    float* xwmu = ws;            ws += (size_t)N * 16;
    float* xwls = ws;            ws += (size_t)N * 16;
    float* dinv = ws;            ws += (size_t)N;
    int*   deg  = (int*)ws;      ws += (size_t)N;
    float* stat = ws;            ws += 192;             // sums[96] + sqs[96]

    const int gM  = cdiv(N, 64);            // GEMM grid (64-row tiles)
    const int gE  = cdiv(E, 256);

    // ======================= GENConv layer 1 (64 -> 48) ====================
    {
        const int d = 48, d2 = 96;
        size_t nd = (size_t)N * d;
        fill_u32<<<cdiv(nd,256),256,0,stream>>>((unsigned*)mmax, 0u, nd);
        fill_f32<<<cdiv(nd,256),256,0,stream>>>(den, 0.f, nd);
        fill_f32<<<cdiv(nd,256),256,0,stream>>>(num, 0.f, nd);
        gemm_wmma_f16<<<gM,128,0,stream>>>(x, w_src1, b_src1, hsrc, N, 64, d, 0);
        edge_msg_max<<<gE,256,0,stream>>>(ea, srcI, dstI, hsrc, w_edg1, b_edg1, mmax, E, d);
        edge_msg_sum<<<gE,256,0,stream>>>(ea, srcI, dstI, hsrc, w_edg1, b_edg1, mmax, den, num, E, d);
        combine_agg<<<cdiv(nd,256),256,0,stream>>>(den, num, hsrc, nd);
        gemm_wmma_f16<<<gM,128,0,stream>>>(den, w_m1a, b_m1a, zt, N, d, d2, 0);
        fill_f32<<<1,256,0,stream>>>(stat, 0.f, 192);
        bn_reduce<<<480,256,0,stream>>>(zt, stat, stat + 96, N, d2);
        bn_apply_relu<<<cdiv((size_t)N*d2,256),256,0,stream>>>(zt, stat, stat + 96, g1, be1, N, d2);
        gemm_wmma_f16<<<gM,128,0,stream>>>(zt, w_m1b, b_m1b, hmid, N, d2, d, 1); // + outer relu
    }

    // ======================= GENConv layer 2 (48 -> 32) ====================
    {
        const int d = 32, d2 = 64;
        size_t nd = (size_t)N * d;
        fill_u32<<<cdiv(nd,256),256,0,stream>>>((unsigned*)mmax, 0u, nd);
        fill_f32<<<cdiv(nd,256),256,0,stream>>>(den, 0.f, nd);
        fill_f32<<<cdiv(nd,256),256,0,stream>>>(num, 0.f, nd);
        gemm_wmma_f16<<<gM,128,0,stream>>>(hmid, w_src2, b_src2, hsrc, N, 48, d, 0);
        edge_msg_max<<<gE,256,0,stream>>>(ea, srcI, dstI, hsrc, w_edg2, b_edg2, mmax, E, d);
        edge_msg_sum<<<gE,256,0,stream>>>(ea, srcI, dstI, hsrc, w_edg2, b_edg2, mmax, den, num, E, d);
        combine_agg<<<cdiv(nd,256),256,0,stream>>>(den, num, hsrc, nd);
        gemm_wmma_f16<<<gM,128,0,stream>>>(den, w_m2a, b_m2a, zt, N, d, d2, 0);
        fill_f32<<<1,256,0,stream>>>(stat, 0.f, 192);
        bn_reduce<<<480,256,0,stream>>>(zt, stat, stat + 96, N, d2);
        bn_apply_relu<<<cdiv((size_t)N*d2,256),256,0,stream>>>(zt, stat, stat + 96, g2, be2, N, d2);
        gemm_wmma_f16<<<gM,128,0,stream>>>(zt, w_m2b, b_m2b, hfin, N, d2, d, 1); // + outer relu
    }

    // ========================= GCN heads (32 -> 16) ========================
    float* out_mu = (float*)d_out;
    float* out_ls = out_mu + (size_t)N * 16;
    gemm_wmma_f16<<<gM,128,0,stream>>>(hfin, w_mu, nullptr, xwmu, N, 32, 16, 0);
    gemm_wmma_f16<<<gM,128,0,stream>>>(hfin, w_ls, nullptr, xwls, N, 32, 16, 0);
    fill_u32<<<cdiv(N,256),256,0,stream>>>((unsigned*)deg, 0u, (size_t)N);
    gcn_deg<<<gE,256,0,stream>>>(dstI, deg, E);
    gcn_dinv<<<cdiv(N,256),256,0,stream>>>(deg, dinv, N);
    gcn_init<<<cdiv((size_t)N*16,256),256,0,stream>>>(xwmu, xwls, dinv, b_mu, b_ls, out_mu, out_ls, N);
    gcn_edge<<<gE,256,0,stream>>>(srcI, dstI, dinv, xwmu, xwls, out_mu, out_ls, E);
}
